// GenBack_64476049048238
// MI455X (gfx1250) — matrix operand order repaired and checked by reading, not compiled, not run on hardware
//
#include <hip/hip_runtime.h>
#include <hip/hip_bf16.h>
#include <math.h>

// ---------------------------------------------------------------------------
// Types for CDNA5 WMMA (wave32, 16x16x32 f16 -> f32)
// ---------------------------------------------------------------------------
typedef _Float16 half_t;
typedef __attribute__((ext_vector_type(8)))  _Float16 v8h;
typedef __attribute__((ext_vector_type(16))) _Float16 v16h;
typedef __attribute__((ext_vector_type(8)))  float    v8f;

// A-matrix 16x32 f16 fragment: lanes 0-15 hold M=0..15, lanes 16-31 the same M
// with the other K half. Per lane: elements 0..7 = K base+0..7 (contiguous),
// elements 8..15 = K base+16..23 (contiguous). Two 16B ds_load_b128.
__device__ __forceinline__ v16h load_a_frag(const half_t* rowK, int kOff) {
  union { v16h v; v8h h[2]; } u;
  u.h[0] = *(const v8h*)(rowK + kOff);
  u.h[1] = *(const v8h*)(rowK + kOff + 16);
  return u.v;
}

// B-matrix 32x16 f16 fragment with weights packed [N][K]: per lane,
// elements 0..7 = K base+0..7, elements 8..15 = K base+16..23 (two global_load_b128).
__device__ __forceinline__ v16h load_b_frag(const half_t* colK) {
  union { v16h v; v8h h[2]; } u;
  u.h[0] = *(const v8h*)(colK);
  u.h[1] = *(const v8h*)(colK + 16);
  return u.v;
}

__device__ __forceinline__ v8f wmma_f16(v16h a, v16h b, v8f c) {
  return __builtin_amdgcn_wmma_f32_16x16x32_f16(false, a, false, b,
                                                (short)0, c, false, false);
}

// ---------------------------------------------------------------------------
// Weight / input packing (fp32 -> f16, WMMA-friendly layouts)
// ---------------------------------------------------------------------------
__global__ __launch_bounds__(256) void pack_z(const float* __restrict__ z,
                                              half_t* __restrict__ zA) {
  int i = blockIdx.x * 256 + threadIdx.x;            // [128][128] K-padded
  if (i >= 128 * 128) return;
  int m = i >> 7, k = i & 127;
  zA[i] = (half_t)(k < 100 ? z[m * 100 + k] : 0.0f);
}

__global__ __launch_bounds__(256) void pack_w1(const float* __restrict__ w1,
                                               half_t* __restrict__ w1p) {
  int i = blockIdx.x * 256 + threadIdx.x;            // [n=8192][k=128]
  if (i >= 8192 * 128) return;
  int n = i >> 7, k = i & 127;
  w1p[i] = (half_t)(k < 100 ? w1[(size_t)k * 8192 + n] : 0.0f);
}

// Wc [Co][Ci][3][3] fp32 -> wp [NpadCout][K=9*Ci] f16 with K = (kh*3+kw)*Ci+ci
__global__ __launch_bounds__(256) void pack_wc(const float* __restrict__ wc,
                                               half_t* __restrict__ wp,
                                               int Co, int Np, int Ci) {
  int K = 9 * Ci;
  int i = blockIdx.x * 256 + threadIdx.x;
  if (i >= Np * K) return;
  int co = i / K, k = i - co * K;
  int khw = k / Ci, ci = k - khw * Ci;
  int kh = khw / 3, kw = khw - kh * 3;
  float v = (co < Co) ? wc[(((size_t)co * Ci + ci) * 3 + kh) * 3 + kw] : 0.0f;
  wp[i] = (half_t)v;
}

// ---------------------------------------------------------------------------
// GEMM: act0[128,8192] = zA[128,128(f16)] x W1p + b1  (WMMA)
// 64 blocks over N; 8 waves/block each own a 16-col strip x all 8 M-tiles.
// ---------------------------------------------------------------------------
__global__ __launch_bounds__(256) void gemm_z_w1(const half_t* __restrict__ zA,
                                                 const half_t* __restrict__ w1p,
                                                 const float*  __restrict__ b1,
                                                 float* __restrict__ out) {
  __shared__ __attribute__((aligned(16))) half_t za[128 * 128];
  const int tid = threadIdx.x;
  for (int i = tid; i < (128 * 128) / 8; i += 256)
    ((v8h*)za)[i] = ((const v8h*)zA)[i];
  __syncthreads();

  const int lane = tid & 31, wave = tid >> 5;
  const int hi = lane >> 4, ml = lane & 15;
  const int ncol = blockIdx.x * 128 + wave * 16 + ml;
  const half_t* wrow = w1p + (size_t)ncol * 128;

  v8f acc[8] = {};
  for (int ks = 0; ks < 4; ++ks) {
    v16h b = load_b_frag(wrow + ks * 32 + hi * 8);
#pragma unroll
    for (int mt = 0; mt < 8; ++mt) {
      v16h a = load_a_frag(za + (mt * 16 + ml) * 128, ks * 32 + hi * 8);
      acc[mt] = wmma_f16(a, b, acc[mt]);
    }
  }
  float bv = b1[ncol];
#pragma unroll
  for (int mt = 0; mt < 8; ++mt)
#pragma unroll
    for (int r = 0; r < 8; ++r)
      out[(size_t)(mt * 16 + hi * 8 + r) * 8192 + ncol] = acc[mt][r] + bv;
}

// ---------------------------------------------------------------------------
// 3x3 conv as implicit GEMM with WMMA. One workgroup per image; whole image
// (with halo, channel-innermost) staged in LDS (<=289KB, fits 320KB WGP LDS).
// Each work unit = one 16-wide N strip x 8 M-tiles: one B fragment load per
// K-step feeds 8 WMMAs (B reuse, GEMM-style).
// ---------------------------------------------------------------------------
template<int CIN, int COUT, int COUT_REAL, int H, int W>
__global__ __launch_bounds__(256) void conv3x3_wmma(
    const half_t* __restrict__ xin,   // [B][CIN][H*W] f16
    const half_t* __restrict__ wp,    // [COUT][9*CIN] f16, (kh,kw,ci)-ordered K
    const float*  __restrict__ bias,  // [COUT_REAL]
    float* __restrict__ out) {        // [B][COUT_REAL][H*W] f32
  extern __shared__ __attribute__((aligned(16))) half_t lds[]; // [(H+2)*(W+2)][CIN]
  constexpr int HW = H * W;
  constexpr int PW = W + 2;
  constexpr int LDSN = (H + 2) * PW * CIN;
  constexpr int MT = HW / 16;          // M tiles (16, 64, 64)
  constexpr int NT = COUT / 16;        // N tiles (8, 4, 1)
  constexpr int MG = MT / 8;           // m-groups of 8 tiles
  constexpr int UNITS = NT * MG;       // 16 / 32 / 8
  constexpr int KSTEPS = (9 * CIN) / 32;

  const int img  = blockIdx.x;
  const int tid  = threadIdx.x;
  const int lane = tid & 31, wave = tid >> 5;
  const int hi = lane >> 4, ml = lane & 15;

  // zero (covers halo), then stage [pixel][channel]
  {
    float* l32 = (float*)lds;
    for (int i = tid; i < LDSN / 2; i += 256) l32[i] = 0.0f;
  }
  __syncthreads();
  {
    const half_t* src = xin + (size_t)img * CIN * HW;
    for (int i = tid; i < CIN * HW; i += 256) {
      int ci = i / HW;
      int s  = i - ci * HW;
      int h  = s / W, w = s - h * W;
      lds[((h + 1) * PW + (w + 1)) * CIN + ci] = src[i];
    }
  }
  __syncthreads();

  for (int u = wave; u < UNITS; u += 8) {
    const int n0 = (u % NT) * 16;
    const int g  = u / NT;
    const half_t* wrow = wp + (size_t)(n0 + ml) * (9 * CIN);

    // halo-centered LDS pixel index for each of this unit's 8 A rows
    int rowbase[8];
#pragma unroll
    for (int j = 0; j < 8; ++j) {
      int m = (g * 8 + j) * 16 + ml;         // A: M = lane&15
      int oh = m / W, ow = m - oh * W;
      rowbase[j] = (oh + 1) * PW + (ow + 1);
    }

    v8f acc[8] = {};
    for (int ks = 0; ks < KSTEPS; ++ks) {
      const int kbase = ks * 32;             // stays inside one (kh,kw) run
      const int khw = kbase / CIN;
      const int kh = khw / 3, kw = khw - kh * 3;
      const int cib = (kbase - khw * CIN) + hi * 8;
      const int roff = (kh - 1) * PW + (kw - 1);
      v16h b = load_b_frag(wrow + kbase + hi * 8);
#pragma unroll
      for (int j = 0; j < 8; ++j) {
        const half_t* arow = lds + (size_t)(rowbase[j] + roff) * CIN;
        v16h a = load_a_frag(arow, cib);
        acc[j] = wmma_f16(a, b, acc[j]);
      }
    }

    const int co = n0 + ml;                  // C/D: N = lane&15
    if (co < COUT_REAL) {
      const float bv = bias[co];
      float* orow = out + ((size_t)img * COUT_REAL + co) * HW;
#pragma unroll
      for (int j = 0; j < 8; ++j) {
        const int mbase = (g * 8 + j) * 16 + hi * 8;   // C/D: M = r + hi*8
        float tmp[8];
#pragma unroll
        for (int r = 0; r < 8; ++r) tmp[r] = acc[j][r] + bv;
        *(float4*)(orow + mbase)     = *(float4*)(tmp);
        *(float4*)(orow + mbase + 4) = *(float4*)(tmp + 4);
      }
    }
  }
}

// ---------------------------------------------------------------------------
// BatchNorm stats (training mode, biased var) -> folded scale/shift per channel
// One block per channel; element (b,c,s) at b*strideB + c*S + s.
// ---------------------------------------------------------------------------
__global__ __launch_bounds__(256) void bn_stats(const float* __restrict__ src,
                                                const float* __restrict__ gamma,
                                                const float* __restrict__ beta,
                                                float* __restrict__ stats,
                                                int B, int S, int strideB,
                                                float eps) {
  __shared__ float s1[256], s2[256];
  const int c = blockIdx.x, tid = threadIdx.x;
  const float* base = src + (size_t)c * S;
  const int total = B * S;
  float sum = 0.0f, sq = 0.0f;
  for (int i = tid; i < total; i += 256) {
    int b = i / S, s = i - b * S;
    float v = base[(size_t)b * strideB + s];
    sum += v; sq += v * v;
  }
  s1[tid] = sum; s2[tid] = sq;
  __syncthreads();
  for (int off = 128; off > 0; off >>= 1) {
    if (tid < off) { s1[tid] += s1[tid + off]; s2[tid] += s2[tid + off]; }
    __syncthreads();
  }
  if (tid == 0) {
    float mean = s1[0] / (float)total;
    float var  = s2[0] / (float)total - mean * mean;
    float sc   = gamma[c] * rsqrtf(var + eps);
    stats[c * 2]     = sc;
    stats[c * 2 + 1] = beta[c] - mean * sc;
  }
}

// BN apply (+optional LeakyReLU 0.2) (+optional nearest-up x`up`), pack to f16
__global__ __launch_bounds__(256) void bn_apply_pack(
    const float* __restrict__ src, const float* __restrict__ stats,
    half_t* __restrict__ dst, int C, int Hin, int Win, int up, int leaky,
    int total) {
  int i = blockIdx.x * 256 + threadIdx.x;
  if (i >= total) return;
  const int Wo = Win * up, Ho = Hin * up;
  const int So = Ho * Wo;
  int s   = i % So;
  int c   = (i / So) % C;
  int img = i / (So * C);
  int oh = s / Wo, ow = s - oh * Wo;
  int sin = (oh / up) * Win + (ow / up);
  float v = src[((size_t)img * C + c) * (Hin * Win) + sin];
  v = v * stats[c * 2] + stats[c * 2 + 1];
  if (leaky && v < 0.0f) v *= 0.2f;
  dst[i] = (half_t)v;
}

// ---------------------------------------------------------------------------
// 200-step LIF (reset-by-subtract) + tanh image. This is the roofline term
// (~630MB of stores): 4 elements per thread -> all stores are b128, coalesced
// across the t-major [200][N] layout.
// ---------------------------------------------------------------------------
__global__ __launch_bounds__(256) void lif_tanh(const float* __restrict__ x,
                                                float* __restrict__ img,
                                                float* __restrict__ spk,
                                                float* __restrict__ mem,
                                                int N) {
  int i4 = (blockIdx.x * 256 + threadIdx.x) * 4;
  if (i4 >= N) return;
  float xv[4], m[4], s[4], t[4];
  *(float4*)xv = *(const float4*)(x + i4);
#pragma unroll
  for (int j = 0; j < 4; ++j) { t[j] = tanhf(xv[j]); m[j] = 0.0f; }
  *(float4*)(img + i4) = *(float4*)t;
  size_t idx = (size_t)i4;
  for (int tt = 0; tt < 200; ++tt) {
#pragma unroll
    for (int j = 0; j < 4; ++j) {
      float reset = (m[j] > 1.0f) ? 1.0f : 0.0f;   // THR = 1
      m[j] = 0.95f * m[j] + xv[j] - reset;         // BETA = 0.95
      s[j] = (m[j] > 1.0f) ? 1.0f : 0.0f;
    }
    *(float4*)(spk + idx) = *(float4*)s;
    *(float4*)(mem + idx) = *(float4*)m;
    idx += (size_t)N;
  }
}

// ---------------------------------------------------------------------------
// Host side
// ---------------------------------------------------------------------------
extern "C" void kernel_launch(void* const* d_in, const int* in_sizes, int n_in,
                              void* d_out, int out_size, void* d_ws,
                              size_t ws_size, hipStream_t stream) {
  (void)in_sizes; (void)n_in; (void)out_size; (void)ws_size;
  const float* z   = (const float*)d_in[0];
  const float* W1  = (const float*)d_in[1];
  const float* b1  = (const float*)d_in[2];
  const float* g1  = (const float*)d_in[3];
  const float* be1 = (const float*)d_in[4];
  const float* g2  = (const float*)d_in[5];
  const float* be2 = (const float*)d_in[6];
  const float* g3  = (const float*)d_in[7];
  const float* be3 = (const float*)d_in[8];
  const float* Wc1 = (const float*)d_in[9];
  const float* bc1 = (const float*)d_in[10];
  const float* Wc2 = (const float*)d_in[11];
  const float* bc2 = (const float*)d_in[12];
  const float* Wc3 = (const float*)d_in[13];
  const float* bc3 = (const float*)d_in[14];

  char* ws = (char*)d_ws;
  size_t off = 0;
  auto take = [&](size_t bytes) -> char* {
    char* p = ws + off;
    off += (bytes + 255) & ~(size_t)255;
    return p;
  };

  half_t* zA   = (half_t*)take((size_t)128 * 128 * 2);
  half_t* w1p  = (half_t*)take((size_t)8192 * 128 * 2);
  float*  act0 = (float*) take((size_t)128 * 8192 * 4);
  float*  st1  = (float*) take(128 * 2 * 4);
  float*  st2  = (float*) take(128 * 2 * 4);
  float*  st3  = (float*) take(64 * 2 * 4);
  half_t* x1   = (half_t*)take((size_t)128 * 128 * 256 * 2);
  half_t* wc1p = (half_t*)take((size_t)128 * 1152 * 2);
  float*  c1o  = (float*) take((size_t)128 * 128 * 256 * 4);
  half_t* x2   = (half_t*)take((size_t)128 * 128 * 1024 * 2);
  half_t* wc2p = (half_t*)take((size_t)64 * 1152 * 2);
  float*  c2o  = (float*) take((size_t)128 * 64 * 1024 * 4);
  half_t* x3   = (half_t*)take((size_t)128 * 64 * 1024 * 2);
  half_t* wc3p = (half_t*)take((size_t)16 * 576 * 2);
  float*  o3   = (float*) take((size_t)128 * 3 * 1024 * 4);

  // pack inputs / weights to f16 WMMA layouts
  pack_z <<<(128 * 128 + 255) / 256, 256, 0, stream>>>(z, zA);
  pack_w1<<<(8192 * 128 + 255) / 256, 256, 0, stream>>>(W1, w1p);
  pack_wc<<<(128 * 1152 + 255) / 256, 256, 0, stream>>>(Wc1, wc1p, 128, 128, 128);
  pack_wc<<<(64  * 1152 + 255) / 256, 256, 0, stream>>>(Wc2, wc2p, 64, 64, 128);
  pack_wc<<<(16  * 576  + 255) / 256, 256, 0, stream>>>(Wc3, wc3p, 3, 16, 64);

  // l1: z @ W1 + b1
  gemm_z_w1<<<64, 256, 0, stream>>>(zA, w1p, b1, act0);

  // BN1 (eps 1e-5) + up x2 -> x1 f16 [128,128,16,16]
  bn_stats<<<128, 256, 0, stream>>>(act0, g1, be1, st1, 128, 64, 8192, 1e-5f);
  bn_apply_pack<<<(128 * 128 * 256) / 256, 256, 0, stream>>>(
      act0, st1, x1, 128, 8, 8, 2, 0, 128 * 128 * 256);

  // conv1 128->128 @16x16 (LDS stage 18*18*128 f16 = 81KB)
  conv3x3_wmma<128, 128, 128, 16, 16>
      <<<128, 256, 18 * 18 * 128 * 2, stream>>>(x1, wc1p, bc1, c1o);

  // BN2 (eps 0.8) + LeakyReLU + up x2 -> x2 f16 [128,128,32,32]
  bn_stats<<<128, 256, 0, stream>>>(c1o, g2, be2, st2, 128, 256, 128 * 256, 0.8f);
  bn_apply_pack<<<(128 * 128 * 1024) / 256, 256, 0, stream>>>(
      c1o, st2, x2, 128, 16, 16, 2, 1, 128 * 128 * 1024);

  // conv2 128->64 @32x32 (LDS stage 34*34*128 f16 = 289KB, fits 320KB WGP LDS)
  conv3x3_wmma<128, 64, 64, 32, 32>
      <<<128, 256, 34 * 34 * 128 * 2, stream>>>(x2, wc2p, bc2, c2o);

  // BN3 (eps 0.8) + LeakyReLU -> x3 f16 [128,64,32,32]
  bn_stats<<<64, 256, 0, stream>>>(c2o, g3, be3, st3, 128, 1024, 64 * 1024, 0.8f);
  bn_apply_pack<<<(128 * 64 * 1024) / 256, 256, 0, stream>>>(
      c2o, st3, x3, 64, 32, 32, 1, 1, 128 * 64 * 1024);

  // conv3 64->3 (N padded to 16) @32x32 (LDS 34*34*64 f16 = 145KB)
  conv3x3_wmma<64, 16, 3, 32, 32>
      <<<128, 256, 34 * 34 * 64 * 2, stream>>>(x3, wc3p, bc3, o3);

  // outputs: img [N], spk_rec [200*N], mem_rec [200*N]
  const int N = 128 * 3 * 32 * 32;
  float* out = (float*)d_out;
  lif_tanh<<<(N / 4 + 255) / 256, 256, 0, stream>>>(o3, out, out + N,
                                                    out + (size_t)N * 201, N);
}
